// VectorModule_78829829751255
// MI455X (gfx1250) — matrix operand order repaired and checked by reading, not compiled, not run on hardware
//
#include <hip/hip_runtime.h>
#include <hip/hip_bf16.h>

// ---------------------------------------------------------------------------
// MI455X (gfx1250) implementation.
// Dominant cost: 7 per-pair filter MLPs (16->16->16 over 1M pairs) ~7.3 GFLOP,
// mapped onto v_wmma_f32_16x16x32_f16 (weights as transposed A-matrix,
// rbf/hidden as B-matrix). All intermediates ~1.3MB => L2 resident; fused so
// the N^2 filter tensors are never materialized.
// Round-2 changes: fast v_rcp/v_rsq transcendentals instead of IEEE divides in
// the hot pair loops; cross-half lane exchange via v_permlanex16 (VALU) rather
// than LDS swizzle, removing s_wait_dscnt stalls between the two WMMAs.
// ---------------------------------------------------------------------------

#define NN    1024
#define EPSF  1e-8f
#define RMAXF 10.0f
#define PIF   3.14159265358979323846f

typedef __attribute__((ext_vector_type(16))) _Float16 v16h;
typedef __attribute__((ext_vector_type(8)))  float    v8f;

struct FiltW { const float *W1, *W2, *b1, *b2; };
struct L0W   { const float *W1, *W2, *b1, *b2; };
struct L1W   { const float *Wmix, *gW1, *gW2, *gb1, *gb2; };

__device__ __forceinline__ float frcp(float x)  { return __builtin_amdgcn_rcpf(x); }
__device__ __forceinline__ float frsq(float x)  { return __builtin_amdgcn_rsqf(x); }
__device__ __forceinline__ float dswish(float x) { return x * frcp(1.0f + __expf(-x)); }
__device__ __forceinline__ float dsigm(float x)  { return frcp(1.0f + __expf(-x)); }

// Cross-half (lane ^ 16) exchange. v_permlanex16_b32 with identity selects
// swaps values between the two 16-lane halves entirely in the VALU pipe.
__device__ __forceinline__ float xor16f(float x) {
#if __has_builtin(__builtin_amdgcn_permlanex16)
  int v = __float_as_int(x);
  v = __builtin_amdgcn_permlanex16(v, v, 0x76543210, 0xfedcba98, true, false);
  return __int_as_float(v);
#else
  return __shfl_xor(x, 16, 32);
#endif
}

// A-fragment = W^T for a 16x16 weight, K padded 16->32 with zeros.
// 16-bit A 16x32 layout (contiguous): lane<16: halves0..15 -> K=0..15 ;
// lane>=16: K=16..31 (zero padding).
__device__ __forceinline__ v16h load_AT16(const float* W, int lane) {
  int m  = lane & 15;
  int rb = (lane < 16) ? 0 : 8;
  v16h a;
#pragma unroll
  for (int h = 0; h < 16; ++h) a[h] = (_Float16)0.0f;
#pragma unroll
  for (int h = 0; h < 16; ++h) a[h] = (lane < 16) ? (_Float16)W[h * 16 + m] : (_Float16)0.0f;  // A[m][k]=W[k][m]
  return a;
}

// One filter MLP on 16 pairs: D = W2^T x swish(W1^T x rbf^T + b1) + b2.
// C-tile of WMMA1 (lane holds column p=lane&15, rows rb..rb+7) becomes the
// B-fragment of WMMA2 via a single xor-16 lane exchange (B 32x16 layout:
// lanes 0..15 hold K=0..15 in halves 0..15; lanes 16..31 = K pad = 0).
__device__ __forceinline__ v8f filter_apply(const v16h& a1, const float* b1,
                                            const v16h& a2, const float* b2,
                                            const v16h& brbf, int lane) {
  v8f zero;
#pragma unroll
  for (int r = 0; r < 8; ++r) zero[r] = 0.0f;
  const int rb = (lane < 16) ? 0 : 8;
  v8f h1 = __builtin_amdgcn_wmma_f32_16x16x32_f16(false, a1, false, brbf,
                                                  (short)0, zero, false, false);
  float t[8];
#pragma unroll
  for (int r = 0; r < 8; ++r) t[r] = dswish(h1[r] + b1[rb + r]);
  v16h bh;
#pragma unroll
  for (int h = 0; h < 16; ++h) bh[h] = (_Float16)0.0f;
#pragma unroll
  for (int r = 0; r < 8; ++r) {
    float oth = xor16f(t[r]);
    bh[r]     = (lane < 16) ? (_Float16)t[r] : (_Float16)0.0f;
    bh[8 + r] = (lane < 16) ? (_Float16)oth : (_Float16)0.0f;
  }
  v8f d = __builtin_amdgcn_wmma_f32_16x16x32_f16(false, a2, false, bh,
                                                 (short)0, zero, false, false);
  v8f o;
#pragma unroll
  for (int r = 0; r < 8; ++r) o[r] = d[r] + b2[rb + r];
  return o;
}

// rbf B-fragment for 16 pairs (column p = lane&15, basis k in halves).
// Each half-wave computes 8 sines; xor-16 exchange assembles lanes 0..15.
__device__ __forceinline__ v16h build_rbf(float nm, float msk, int lane) {
  const int rb = (lane < 16) ? 0 : 8;
  float rinv = msk * frcp(nm + EPSF);
  float my[8];
#pragma unroll
  for (int b = 0; b < 8; ++b) {
    float f = (float)(rb + b + 1) * (PIF / RMAXF);
    my[b] = __sinf(nm * f) * rinv;
  }
  v16h br;
#pragma unroll
  for (int h = 0; h < 16; ++h) br[h] = (_Float16)0.0f;
#pragma unroll
  for (int b = 0; b < 8; ++b) {
    float oth = xor16f(my[b]);
    br[b]     = (lane < 16) ? (_Float16)my[b] : (_Float16)0.0f;
    br[8 + b] = (lane < 16) ? (_Float16)oth : (_Float16)0.0f;
  }
  return br;
}

// ---------- prep: time MLP + v0 = concat(feat0, tvec) ----------------------
__global__ void __launch_bounds__(256) k_prep(const float* __restrict__ feat0,
                                              const float* __restrict__ tW1,
                                              const float* __restrict__ tb1,
                                              const float* __restrict__ tW2,
                                              const float* __restrict__ tb2,
                                              float* __restrict__ v0) {
  int t = blockIdx.x * blockDim.x + threadIdx.x;
  if (t >= NN * 16) return;
  int i = t >> 4, c = t & 15;
  float val;
  if (c < 8) {
    val = feat0[i * 8 + c];
  } else {
    int cc = c - 8;
    float acc = tb2[cc];
#pragma unroll
    for (int k = 0; k < 8; ++k) {
      float h = 1.0f * tW1[k] + tb1[k];  // T = 1.0, W1 shape [1,8]
      acc += dswish(h) * tW2[k * 8 + cc];
    }
    val = acc;
  }
  v0[i * 16 + c] = val;
}

// ---------- layer-0 pair pass: 2 filters, c00 + c01 ------------------------
__global__ void __launch_bounds__(256) k_pairs0(
    const float* __restrict__ pos, const float* __restrict__ v0,
    FiltW f00, FiltW f01,
    float* __restrict__ c00o, float* __restrict__ c01o) {
  __shared__ float sa[16 * 16 * 4];
  const int tid = threadIdx.x;
  for (int idx = tid; idx < 16 * 16 * 4; idx += 256) sa[idx] = 0.0f;
  __syncthreads();

  const int lane = tid & 31;
  const int wave = tid >> 5;
  const int il   = lane & 15;
  const int rb   = (lane < 16) ? 0 : 8;
  const int i    = blockIdx.x * 16 + il;
  const float pix = pos[i * 3 + 0], piy = pos[i * 3 + 1], piz = pos[i * 3 + 2];

  const v16h a1_00 = load_AT16(f00.W1, lane), a2_00 = load_AT16(f00.W2, lane);
  const v16h a1_01 = load_AT16(f01.W1, lane), a2_01 = load_AT16(f01.W2, lane);

  float acc00[8], acc01[8][3];
#pragma unroll
  for (int r = 0; r < 8; ++r) { acc00[r] = 0.f; acc01[r][0] = acc01[r][1] = acc01[r][2] = 0.f; }

  for (int j = wave; j < NN; j += 8) {
    __builtin_prefetch(&v0[((j + 8) & (NN - 1)) * 16], 0, 1);
    float pjx = pos[j * 3 + 0], pjy = pos[j * 3 + 1], pjz = pos[j * 3 + 2];
    float rx = pix - pjx, ry = piy - pjy, rz = piz - pjz;
    float n2 = rx * rx + ry * ry + rz * rz + EPSF;
    float inv = frsq(n2);        // 1/sqrt(n2)
    float nrm = n2 * inv;        // sqrt(n2)
    float ux = rx * inv, uy = ry * inv, uz = rz * inv;
    float msk = (i != j) ? 1.0f : 0.0f;
    float nm = nrm * msk;

    v16h brbf = build_rbf(nm, msk, lane);
    v8f R00 = filter_apply(a1_00, f00.b1, a2_00, f00.b2, brbf, lane);
    v8f R01 = filter_apply(a1_01, f01.b1, a2_01, f01.b2, brbf, lane);

    const float* vj = v0 + j * 16 + rb;
#pragma unroll
    for (int r = 0; r < 8; ++r) {
      float vv = vj[r];
      float w00 = R00[r] * msk, w01 = R01[r] * msk;
      acc00[r] += w00 * vv;
      float t = w01 * vv;
      acc01[r][0] += t * ux; acc01[r][1] += t * uy; acc01[r][2] += t * uz;
    }
  }

#pragma unroll
  for (int r = 0; r < 8; ++r) {
    int c = rb + r;
    float* p = &sa[(il * 16 + c) * 4];
    atomicAdd(p + 0, acc00[r]);
    atomicAdd(p + 1, acc01[r][0]);
    atomicAdd(p + 2, acc01[r][1]);
    atomicAdd(p + 3, acc01[r][2]);
  }
  __syncthreads();
  for (int idx = tid; idx < 256; idx += 256) {
    int ii = blockIdx.x * 16 + (idx >> 4);
    int c  = idx & 15;
    const float* p = &sa[idx * 4];
    c00o[ii * 16 + c] = p[0];
    c01o[(ii * 16 + c) * 3 + 0] = p[1];
    c01o[(ii * 16 + c) * 3 + 1] = p[2];
    c01o[(ii * 16 + c) * 3 + 2] = p[3];
  }
}

// ---------- layer-0 node MLPs: L0 (scalar path), L1 (vector path) ----------
__global__ void __launch_bounds__(256) k_node0(const float* __restrict__ c00,
                                               const float* __restrict__ c01,
                                               L0W m0, L1W m1,
                                               float* __restrict__ L0o,
                                               float* __restrict__ L1o) {
  int t = blockIdx.x * blockDim.x + threadIdx.x;
  if (t >= NN * 16) return;
  int i = t >> 4, c = t & 15;
  const float* x = c00 + i * 16;
  float acc = m0.b2[c];
  for (int k = 0; k < 16; ++k) {
    float h = m0.b1[k];
#pragma unroll 4
    for (int q = 0; q < 16; ++q) h += x[q] * m0.W1[q * 16 + k];
    acc += dswish(h) * m0.W2[k * 16 + c];
  }
  L0o[i * 16 + c] = acc;

  const float* v = c01 + i * 48;
  float nq[16];
  for (int q = 0; q < 16; ++q) {
    float a = v[q * 3], b = v[q * 3 + 1], d = v[q * 3 + 2];
    nq[q] = sqrtf(a * a + b * b + d * d + EPSF);
  }
  float g = m1.gb2[c];
  for (int k = 0; k < 16; ++k) {
    float h = m1.gb1[k];
    for (int q = 0; q < 16; ++q) h += nq[q] * m1.gW1[q * 16 + k];
    g += dswish(h) * m1.gW2[k * 16 + c];
  }
  float sg = dsigm(g);
#pragma unroll
  for (int m = 0; m < 3; ++m) {
    float mx = 0.0f;
    for (int q = 0; q < 16; ++q) mx += v[q * 3 + m] * m1.Wmix[q * 16 + c];
    L1o[(i * 16 + c) * 3 + m] = mx * sg;
  }
}

// ---------- layer-1 pair pass: 5 filters, c00/c11d/c01/c10/c11c ------------
__global__ void __launch_bounds__(256) k_pairs1(
    const float* __restrict__ pos, const float* __restrict__ v0,
    const float* __restrict__ L1in,
    FiltW f00, FiltW f01, FiltW f10, FiltW f11d, FiltW f11c,
    float* __restrict__ conv0, float* __restrict__ conv1) {
  __shared__ float sa[16 * 16 * 11];
  const int tid = threadIdx.x;
  for (int idx = tid; idx < 16 * 16 * 11; idx += 256) sa[idx] = 0.0f;
  __syncthreads();

  const int lane = tid & 31;
  const int wave = tid >> 5;
  const int il   = lane & 15;
  const int rb   = (lane < 16) ? 0 : 8;
  const int i    = blockIdx.x * 16 + il;
  const float pix = pos[i * 3 + 0], piy = pos[i * 3 + 1], piz = pos[i * 3 + 2];

  const v16h a1_00 = load_AT16(f00.W1, lane),  a2_00 = load_AT16(f00.W2, lane);
  const v16h a1_01 = load_AT16(f01.W1, lane),  a2_01 = load_AT16(f01.W2, lane);
  const v16h a1_10 = load_AT16(f10.W1, lane),  a2_10 = load_AT16(f10.W2, lane);
  const v16h a1_d  = load_AT16(f11d.W1, lane), a2_d  = load_AT16(f11d.W2, lane);
  const v16h a1_c  = load_AT16(f11c.W1, lane), a2_c  = load_AT16(f11c.W2, lane);

  float a00[8], ad[8], a01[8][3], a10[8][3], ac[8][3];
#pragma unroll
  for (int r = 0; r < 8; ++r) {
    a00[r] = 0.f; ad[r] = 0.f;
    a01[r][0] = a01[r][1] = a01[r][2] = 0.f;
    a10[r][0] = a10[r][1] = a10[r][2] = 0.f;
    ac[r][0] = ac[r][1] = ac[r][2] = 0.f;
  }

  for (int j = wave; j < NN; j += 8) {
    __builtin_prefetch(&L1in[((j + 8) & (NN - 1)) * 48], 0, 1);
    float pjx = pos[j * 3 + 0], pjy = pos[j * 3 + 1], pjz = pos[j * 3 + 2];
    float rx = pix - pjx, ry = piy - pjy, rz = piz - pjz;
    float n2 = rx * rx + ry * ry + rz * rz + EPSF;
    float inv = frsq(n2);
    float nrm = n2 * inv;
    float ux = rx * inv, uy = ry * inv, uz = rz * inv;
    float msk = (i != j) ? 1.0f : 0.0f;
    float nm = nrm * msk;

    v16h brbf = build_rbf(nm, msk, lane);
    v8f R00 = filter_apply(a1_00, f00.b1,  a2_00, f00.b2,  brbf, lane);
    v8f R01 = filter_apply(a1_01, f01.b1,  a2_01, f01.b2,  brbf, lane);
    v8f R10 = filter_apply(a1_10, f10.b1,  a2_10, f10.b2,  brbf, lane);
    v8f Rd  = filter_apply(a1_d,  f11d.b1, a2_d,  f11d.b2, brbf, lane);
    v8f Rc  = filter_apply(a1_c,  f11c.b1, a2_c,  f11c.b2, brbf, lane);

    const float* vj = v0 + j * 16 + rb;
    const float* Lj = L1in + (j * 16 + rb) * 3;
#pragma unroll
    for (int r = 0; r < 8; ++r) {
      float vv = vj[r];
      float Lx = Lj[r * 3 + 0], Ly = Lj[r * 3 + 1], Lz = Lj[r * 3 + 2];
      float w00 = R00[r] * msk, w01 = R01[r] * msk, w10 = R10[r] * msk;
      float wd = Rd[r] * msk, wc = Rc[r] * msk;
      a00[r] += w00 * vv;
      ad[r]  += wd * (ux * Lx + uy * Ly + uz * Lz);
      float t = w01 * vv;
      a01[r][0] += t * ux; a01[r][1] += t * uy; a01[r][2] += t * uz;
      a10[r][0] += w10 * Lx; a10[r][1] += w10 * Ly; a10[r][2] += w10 * Lz;
      ac[r][0] += wc * (uy * Lz - uz * Ly);
      ac[r][1] += wc * (uz * Lx - ux * Lz);
      ac[r][2] += wc * (ux * Ly - uy * Lx);
    }
  }

#pragma unroll
  for (int r = 0; r < 8; ++r) {
    int c = rb + r;
    float* p = &sa[(il * 16 + c) * 11];
    atomicAdd(p + 0, a00[r]);
    atomicAdd(p + 1, ad[r]);
    atomicAdd(p + 2, a01[r][0]); atomicAdd(p + 3, a01[r][1]); atomicAdd(p + 4, a01[r][2]);
    atomicAdd(p + 5, a10[r][0]); atomicAdd(p + 6, a10[r][1]); atomicAdd(p + 7, a10[r][2]);
    atomicAdd(p + 8, ac[r][0]);  atomicAdd(p + 9, ac[r][1]);  atomicAdd(p + 10, ac[r][2]);
  }
  __syncthreads();
  for (int idx = tid; idx < 256; idx += 256) {
    int ii = blockIdx.x * 16 + (idx >> 4);
    int c  = idx & 15;
    const float* p = &sa[idx * 11];
    conv0[ii * 32 + c]      = p[0];
    conv0[ii * 32 + 16 + c] = p[1];
#pragma unroll
    for (int m = 0; m < 3; ++m) {
      conv1[(ii * 48 + c) * 3 + m]      = p[2 + m];
      conv1[(ii * 48 + 16 + c) * 3 + m] = p[5 + m];
      conv1[(ii * 48 + 32 + c) * 3 + m] = p[8 + m];
    }
  }
}

// ---------- final node MLPs + outputs + per-node logp ----------------------
__global__ void __launch_bounds__(256) k_final(const float* __restrict__ pos,
                                               const float* __restrict__ outy,
                                               const float* __restrict__ conv0,
                                               const float* __restrict__ conv1,
                                               L0W m0, L1W m1,
                                               float* __restrict__ dout,
                                               float* __restrict__ lp) {
  int i = blockIdx.x * blockDim.x + threadIdx.x;
  if (i >= NN) return;
  // f0 = mlp_L0(conv0[i,:,0]) : 32 -> 32 -> 1 ; sigma = exp(f0)
  const float* x = conv0 + i * 32;
  float f0 = m0.b2[0];
  for (int k = 0; k < 32; ++k) {
    float h = m0.b1[k];
    for (int q = 0; q < 32; ++q) h += x[q] * m0.W1[q * 32 + k];
    f0 += dswish(h) * m0.W2[k];  // W2 [32,1]
  }
  float sg = __expf(f0);
  // f1 = mlp_L1(conv1[i]) ; translations = f1[:,1,:]
  const float* v = conv1 + i * 48 * 3;
  float nq[48];
  for (int q = 0; q < 48; ++q) {
    float a = v[q * 3], b = v[q * 3 + 1], c2 = v[q * 3 + 2];
    nq[q] = sqrtf(a * a + b * b + c2 * c2 + EPSF);
  }
  float g = m1.gb2[1];
  for (int k = 0; k < 32; ++k) {
    float h = m1.gb1[k];
    for (int q = 0; q < 48; ++q) h += nq[q] * m1.gW1[q * 32 + k];
    g += dswish(h) * m1.gW2[k * 2 + 1];  // gW2 [32,2], output channel 1
  }
  float gate = dsigm(g);

  const float C0 = 0.9189385332046727f;  // 0.5*log(2*pi)
  float logs = __logf(sg);
  float isg  = frcp(sg);
  float lpi = 0.0f;
#pragma unroll
  for (int m = 0; m < 3; ++m) {
    float mx = 0.0f;
    for (int q = 0; q < 48; ++q) mx += v[q * 3 + m] * m1.Wmix[q * 2 + 1];  // Wmix [48,2]
    float tr = mx * gate;
    float mu = pos[i * 3 + m] + tr;
    float z = (outy[i * 3 + m] - mu) * isg;
    lpi += -0.5f * z * z - logs - C0;
    dout[i * 3 + m]               = outy[i * 3 + m];   // out_y
    dout[3073 + i * 3 + m]        = mu;                // mu
    dout[3073 + 3072 + i * 3 + m] = sg;                // sigmas (exp(f0) repeated)
    dout[3073 + 6144 + i * 3 + m] = tr;                // translations
  }
  lp[i] = lpi;
}

// ---------- deterministic logp reduction -----------------------------------
__global__ void __launch_bounds__(256) k_logp(const float* __restrict__ lp,
                                              float* __restrict__ dout) {
  __shared__ float s[256];
  int t = threadIdx.x;
  float a = 0.0f;
  for (int k = t; k < NN; k += 256) a += lp[k];
  s[t] = a;
  __syncthreads();
  for (int off = 128; off > 0; off >>= 1) {
    if (t < off) s[t] += s[t + off];
    __syncthreads();
  }
  if (t == 0) dout[3072] = s[0];
}

// ---------------------------------------------------------------------------
// Input order: jax-style tree flatten (alphabetical dict keys at each level):
// 0 feat0, 1 in_y, 2 out_y, then params.l0 (f00:W1,W2,b1,b2; f01; m0:W1,W2,b1,b2;
// m1:Wmix,gW1,gW2,gb1,gb2), params.l1 (f00,f01,f10,f11c,f11d; m0; m1), params.time.
// ---------------------------------------------------------------------------
extern "C" void kernel_launch(void* const* d_in, const int* in_sizes, int n_in,
                              void* d_out, int out_size, void* d_ws, size_t ws_size,
                              hipStream_t stream) {
  auto F = [&](int k) { return (const float*)d_in[k]; };
  const float* feat0 = F(0);
  const float* in_y  = F(1);
  const float* out_y = F(2);
  FiltW l0f00{F(3), F(4), F(5), F(6)};
  FiltW l0f01{F(7), F(8), F(9), F(10)};
  L0W   l0m0{F(11), F(12), F(13), F(14)};
  L1W   l0m1{F(15), F(16), F(17), F(18), F(19)};
  FiltW l1f00{F(20), F(21), F(22), F(23)};
  FiltW l1f01{F(24), F(25), F(26), F(27)};
  FiltW l1f10{F(28), F(29), F(30), F(31)};
  FiltW l1f11c{F(32), F(33), F(34), F(35)};
  FiltW l1f11d{F(36), F(37), F(38), F(39)};
  L0W   l1m0{F(40), F(41), F(42), F(43)};
  L1W   l1m1{F(44), F(45), F(46), F(47), F(48)};
  const float* tW1 = F(49);
  const float* tW2 = F(50);
  const float* tb1 = F(51);
  const float* tb2 = F(52);

  float* w = (float*)d_ws;
  float* v0_0  = w;            // N*16      = 16384
  float* c00_0 = w + 16384;    // N*16      = 16384
  float* c01_0 = w + 32768;    // N*16*3    = 49152
  float* L0b   = w + 81920;    // N*16      = 16384
  float* L1b   = w + 98304;    // N*16*3    = 49152
  float* conv0 = w + 147456;   // N*32      = 32768
  float* conv1 = w + 180224;   // N*48*3    = 147456
  float* lpbuf = w + 327680;   // N         = 1024

  k_prep<<<64, 256, 0, stream>>>(feat0, tW1, tb1, tW2, tb2, v0_0);
  k_pairs0<<<64, 256, 0, stream>>>(in_y, v0_0, l0f00, l0f01, c00_0, c01_0);
  k_node0<<<64, 256, 0, stream>>>(c00_0, c01_0, l0m0, l0m1, L0b, L1b);
  k_pairs1<<<64, 256, 0, stream>>>(in_y, L0b, L1b, l1f00, l1f01, l1f10, l1f11d,
                                   l1f11c, conv0, conv1);
  k_final<<<4, 256, 0, stream>>>(in_y, out_y, conv0, conv1, l1m0, l1m1,
                                 (float*)d_out, lpbuf);
  k_logp<<<1, 256, 0, stream>>>(lpbuf, (float*)d_out);
}
